// MolGNN_74414603370893
// MI455X (gfx1250) — matrix operand order repaired and checked by reading, not compiled, not run on hardware
//
#include <hip/hip_runtime.h>
#include <hip/hip_bf16.h>
#include <math.h>

// ---- problem constants (match reference) ----
#define H     300     // hidden
#define HP    320     // hidden padded to mult of 32 (WMMA K)
#define NEW   60      // per-table embedding width
#define OUTD  1536
#define KN    544     // 9*60=540 padded to mult of 32
#define KE    192     // 3*60=180 padded to mult of 32
#define NP304 304     // 300 padded to mult of 16 (WMMA N tiles)
#define ATTH  150
#define ATTHP 160

typedef _Float16 half8 __attribute__((ext_vector_type(8)));
typedef _Float16 v16h  __attribute__((ext_vector_type(16)));
typedef float    v8f   __attribute__((ext_vector_type(8)));

// =======================================================================
// WMMA GEMM: C[M,Nout] = epilogue(A[M,Kp](f16) @ B[Kp,Nout] + bias)
// B supplied transposed+padded: BT[Npad, Kp] f16, Npad = ceil16(Nout)*16.
// One wave computes one 16x16 output tile; K loop in steps of 32 using
// v_wmma_f32_16x16x32_f16.
// modes: 0: f32 = acc+bias
//        1: f16 = relu(acc+bias)
//        2: f16 = relu(bn_a*(acc+bias)+bn_b)
//        3: f32 = relu(bn_a*(acc+bias)+bn_b) + resid
//        4: f16 = acc+bias
// =======================================================================
__global__ __launch_bounds__(256)
void wmma_gemm(const _Float16* __restrict__ A, int lda,
               const _Float16* __restrict__ BT, int ldb,
               const float* __restrict__ bias,
               const float* __restrict__ bn_a, const float* __restrict__ bn_b,
               const float* __restrict__ resid, int ldr,
               float* __restrict__ Cf, _Float16* __restrict__ Ch, int ldc,
               int M, int Nout, int Kp, int tilesM, int tilesN, int mode)
{
    int wid  = blockIdx.x * (blockDim.x >> 5) + (threadIdx.x >> 5);
    int lane = threadIdx.x & 31;
    int tM = wid / tilesN;
    int tN = wid - tM * tilesN;
    if (tM >= tilesM) return;

    int r  = lane & 15;
    int hi = lane >> 4;
    int arow = tM * 16 + r;
    if (arow >= M) arow = M - 1;           // clamp (stores are guarded)
    int brow = tN * 16 + r;                // BT row == output column (padded)

    const _Float16* Ap = A  + (size_t)arow * lda;
    const _Float16* Bp = BT + (size_t)brow * ldb;
    const int ka = hi ? 8  : 0;            // A frag K pattern: {0..7,16..23}/{8..15,24..31}
    const int kb = hi ? 8  : 0;            // B frag K pattern: {0..7,16..23}/{8..15,24..31}

    v8f acc = {};
    for (int k0 = 0; k0 < Kp; k0 += 32) {
        half8 a0 = *(const half8*)(Ap + k0 + ka);
        half8 a1 = *(const half8*)(Ap + k0 + ka + 16);
        half8 b0 = *(const half8*)(Bp + k0 + kb);
        half8 b1 = *(const half8*)(Bp + k0 + kb + 16);
        v16h av, bv;
#pragma unroll
        for (int i = 0; i < 8; ++i) {
            av[i] = a0[i]; av[i + 8] = a1[i];
            bv[i] = b0[i]; bv[i + 8] = b1[i];
        }
        acc = __builtin_amdgcn_wmma_f32_16x16x32_f16(
                  false, av, false, bv, (short)0, acc, false, false);
    }

    int col = tN * 16 + r;
    if (col >= Nout) return;
    float bsv = bias ? bias[col] : 0.0f;
    float sA = 0.0f, sB = 0.0f;
    if (mode == 2 || mode == 3) { sA = bn_a[col]; sB = bn_b[col]; }
    int rowbase = tM * 16 + (hi ? 8 : 0);
#pragma unroll
    for (int i = 0; i < 8; ++i) {
        int row = rowbase + i;
        if (row >= M) continue;
        float v = acc[i] + bsv;
        if (mode == 2 || mode == 3) v = v * sA + sB;
        if (mode == 1 || mode == 2 || mode == 3) v = fmaxf(v, 0.0f);
        if (mode == 3) v += resid[(size_t)row * ldr + col];
        if (mode == 0 || mode == 3) Cf[(size_t)row * ldc + col] = v;
        else                        Ch[(size_t)row * ldc + col] = (_Float16)v;
    }
}

// ---- weight pack: BT[n,k] = W[k,n] (f32 row-major [K,Nout]) -> f16, zero pad
__global__ void pack_wT(const float* __restrict__ W, _Float16* __restrict__ BT,
                        int K, int Nout, int Kp, int Np)
{
    size_t idx = (size_t)blockIdx.x * blockDim.x + threadIdx.x;
    size_t total = (size_t)Np * Kp;
    if (idx >= total) return;
    int n = (int)(idx / Kp), k = (int)(idx % Kp);
    float v = (n < Nout && k < K) ? W[(size_t)k * Nout + n] : 0.0f;
    BT[idx] = (_Float16)v;
}

// ---- BN affine fold: a = g*rsqrt(v+eps), b' = b - m*a
__global__ void bn_prep(const float* __restrict__ g, const float* __restrict__ b,
                        const float* __restrict__ m, const float* __restrict__ v,
                        float* __restrict__ a, float* __restrict__ bb, int n)
{
    int i = blockIdx.x * blockDim.x + threadIdx.x;
    if (i >= n) return;
    float av = g[i] * rsqrtf(v[i] + 1e-5f);
    a[i] = av; bb[i] = b[i] - m[i] * av;
}

// ---- embedding gathers (concat of table rows), zero-padded K, f16 out
__global__ void node_gather(const int* __restrict__ x,
                            const float* t0, const float* t1, const float* t2,
                            const float* t3, const float* t4, const float* t5,
                            const float* t6, const float* t7, const float* t8,
                            _Float16* __restrict__ Xn, int N)
{
    size_t idx = (size_t)blockIdx.x * blockDim.x + threadIdx.x;
    if (idx >= (size_t)N * KN) return;
    int i = (int)(idx / KN), j = (int)(idx % KN);
    float v = 0.0f;
    if (j < 9 * NEW) {
        int t = j / NEW, c = j - t * NEW;
        const float* tabs[9] = {t0,t1,t2,t3,t4,t5,t6,t7,t8};
        int id = x[(size_t)i * 9 + t];
        v = tabs[t][(size_t)id * NEW + c];
    }
    Xn[idx] = (_Float16)v;
}

__global__ void edge_gather(const int* __restrict__ ea,
                            const float* t0, const float* t1, const float* t2,
                            _Float16* __restrict__ Xe, int E)
{
    size_t idx = (size_t)blockIdx.x * blockDim.x + threadIdx.x;
    if (idx >= (size_t)E * KE) return;
    int i = (int)(idx / KE), j = (int)(idx % KE);
    float v = 0.0f;
    if (j < 3 * NEW) {
        int t = j / NEW, c = j - t * NEW;
        const float* tabs[3] = {t0,t1,t2};
        int id = ea[(size_t)i * 3 + t];
        v = tabs[t][(size_t)id * NEW + c];
    }
    Xe[idx] = (_Float16)v;
}

// ---- misc elementwise / scatter kernels ----
__global__ void fill_f32(float* __restrict__ p, float v, size_t n)
{
    size_t i = (size_t)blockIdx.x * blockDim.x + threadIdx.x;
    if (i < n) p[i] = v;
}

__global__ void add_vnode(float* __restrict__ h, const float* __restrict__ vh,
                          const int* __restrict__ batch, int N)
{
    size_t idx = (size_t)blockIdx.x * blockDim.x + threadIdx.x;
    if (idx >= (size_t)N * H) return;
    int i = (int)(idx / H), c = (int)(idx % H);
    h[idx] += vh[(size_t)batch[i] * H + c];
}

__global__ void msg_scatter(const float* __restrict__ h, const _Float16* __restrict__ ef,
                            const int* __restrict__ src, const int* __restrict__ dst,
                            float* __restrict__ agg, int E)
{
    size_t idx = (size_t)blockIdx.x * blockDim.x + threadIdx.x;
    if (idx >= (size_t)E * H) return;
    int e = (int)(idx / H), c = (int)(idx % H);
    float m = h[(size_t)src[e] * H + c] + (float)ef[idx];
    if (m > 0.0f) atomicAdd(&agg[(size_t)dst[e] * H + c], m);
}

// o[M,HP] f16 = pad( a[M,H] (+ b[M,H]) )
__global__ void to_half_pad(const float* __restrict__ a, const float* __restrict__ b,
                            _Float16* __restrict__ o, int M)
{
    size_t idx = (size_t)blockIdx.x * blockDim.x + threadIdx.x;
    if (idx >= (size_t)M * HP) return;
    int i = (int)(idx / HP), k = (int)(idx % HP);
    float v = 0.0f;
    if (k < H) {
        v = a[(size_t)i * H + k];
        if (b) v += b[(size_t)i * H + k];
    }
    o[idx] = (_Float16)v;
}

__global__ void init_vh(float* __restrict__ vh, const float* __restrict__ vemb, int G)
{
    size_t idx = (size_t)blockIdx.x * blockDim.x + threadIdx.x;
    if (idx >= (size_t)G * H) return;
    vh[idx] = vemb[idx % H];
}

__global__ void seg_sum_h(const float* __restrict__ h, const int* __restrict__ batch,
                          float* __restrict__ vt, int N)
{
    size_t idx = (size_t)blockIdx.x * blockDim.x + threadIdx.x;
    if (idx >= (size_t)N * H) return;
    int i = (int)(idx / H), c = (int)(idx % H);
    atomicAdd(&vt[(size_t)batch[i] * H + c], h[idx]);
}

__global__ void att_dot(const _Float16* __restrict__ s1, const float* __restrict__ w2,
                        const float* __restrict__ b2, float* __restrict__ s, int N)
{
    int i = blockIdx.x * blockDim.x + threadIdx.x;
    if (i >= N) return;
    float acc = b2[0];
    for (int k = 0; k < ATTH; ++k)
        acc += (float)s1[(size_t)i * ATTHP + k] * w2[k];
    s[i] = acc;
}

__device__ inline void atomicMaxF(float* addr, float val)
{
    if (val >= 0.0f) atomicMax((int*)addr, __float_as_int(val));
    else             atomicMin((unsigned int*)addr, __float_as_uint(val));
}

__global__ void seg_max(const float* __restrict__ s, const int* __restrict__ batch,
                        float* __restrict__ smax, int N)
{
    int i = blockIdx.x * blockDim.x + threadIdx.x;
    if (i >= N) return;
    atomicMaxF(&smax[batch[i]], s[i]);
}

__global__ void exp_den(const float* __restrict__ s, const int* __restrict__ batch,
                        const float* __restrict__ smax, float* __restrict__ ex,
                        float* __restrict__ den, int N)
{
    int i = blockIdx.x * blockDim.x + threadIdx.x;
    if (i >= N) return;
    int b = batch[i];
    float e = expf(s[i] - smax[b]);
    ex[i] = e;
    atomicAdd(&den[b], e);
}

__global__ void pool_weighted(const float* __restrict__ h, const float* __restrict__ ex,
                              const float* __restrict__ den, const int* __restrict__ batch,
                              float* __restrict__ g, int N)
{
    size_t idx = (size_t)blockIdx.x * blockDim.x + threadIdx.x;
    if (idx >= (size_t)N * H) return;
    int i = (int)(idx / H), c = (int)(idx % H);
    int b = batch[i];
    float w = ex[i] / (den[b] + 1e-16f);
    atomicAdd(&g[(size_t)b * H + c], h[idx] * w);
}

__global__ void normalize_rows(const float* __restrict__ in, float* __restrict__ out, int cols)
{
    __shared__ float red[256];
    int row = blockIdx.x;
    float ss = 0.0f;
    for (int c = threadIdx.x; c < cols; c += blockDim.x) {
        float v = in[(size_t)row * cols + c];
        ss += v * v;
    }
    red[threadIdx.x] = ss;
    __syncthreads();
    for (int s = 128; s > 0; s >>= 1) {
        if ((int)threadIdx.x < s) red[threadIdx.x] += red[threadIdx.x + s];
        __syncthreads();
    }
    float inv = 1.0f / fmaxf(sqrtf(red[0]), 1e-12f);
    for (int c = threadIdx.x; c < cols; c += blockDim.x)
        out[(size_t)row * cols + c] = in[(size_t)row * cols + c] * inv;
}

// =======================================================================
extern "C" void kernel_launch(void* const* d_in, const int* in_sizes, int n_in,
                              void* d_out, int out_size, void* d_ws, size_t ws_size,
                              hipStream_t stream)
{
    (void)n_in; (void)ws_size;
    // ---- inputs (setup_inputs dict order, lists flattened) ----
    const int* x     = (const int*)d_in[0];
    const int* eidx  = (const int*)d_in[1];
    const int* eattr = (const int*)d_in[2];
    const int* batch = (const int*)d_in[3];
    const float* nt[9]; for (int i = 0; i < 9; ++i) nt[i] = (const float*)d_in[5 + i];
    const float* et[3]; for (int i = 0; i < 3; ++i) et[i] = (const float*)d_in[14 + i];
    const float* node_proj_w = (const float*)d_in[17];
    const float* node_proj_b = (const float*)d_in[18];
    const float* edge_proj_w = (const float*)d_in[19];
    const float* edge_proj_b = (const float*)d_in[20];
    const float* v_emb       = (const float*)d_in[21];
    const float* conv_w1 = (const float*)d_in[22];
    const float* conv_b1 = (const float*)d_in[23];
    const float* conv_w2 = (const float*)d_in[24];
    const float* conv_b2 = (const float*)d_in[25];
    const float* bn_g = (const float*)d_in[26];
    const float* bn_b = (const float*)d_in[27];
    const float* bn_m = (const float*)d_in[28];
    const float* bn_v = (const float*)d_in[29];
    const float* vn_w1 = (const float*)d_in[30];
    const float* vn_b1 = (const float*)d_in[31];
    const float* vn_bn_g = (const float*)d_in[32];
    const float* vn_bn_b = (const float*)d_in[33];
    const float* vn_bn_m = (const float*)d_in[34];
    const float* vn_bn_v = (const float*)d_in[35];
    const float* vn_w2 = (const float*)d_in[36];
    const float* vn_b2 = (const float*)d_in[37];
    const float* att_w1 = (const float*)d_in[38];
    const float* att_b1 = (const float*)d_in[39];
    const float* att_w2 = (const float*)d_in[40];
    const float* att_b2 = (const float*)d_in[41];
    const float* proj_w = (const float*)d_in[42];
    const float* proj_b = (const float*)d_in[43];

    const int N = in_sizes[0] / 9;
    const int E = in_sizes[1] / 2;
    const int G = out_size / OUTD;
    const int L = in_sizes[22] / (H * H);
    const int* src = eidx;
    const int* dst = eidx + E;

    // ---- workspace carve ----
    char* wbase = (char*)d_ws;
    size_t off = 0;
    auto wsAlloc = [&](size_t bytes) -> void* {
        void* p = wbase + off;
        off += (bytes + 255) & ~(size_t)255;
        return p;
    };
    float*    h     = (float*)   wsAlloc((size_t)N * H * 4);
    _Float16* efeat = (_Float16*)wsAlloc((size_t)E * H * 2);
    float*    vh    = (float*)   wsAlloc((size_t)G * H * 4);
    float*    vt    = (float*)   wsAlloc((size_t)G * H * 4);
    float*    sb    = (float*)   wsAlloc((size_t)N * 4);
    float*    exb   = (float*)   wsAlloc((size_t)N * 4);
    float*    smax  = (float*)   wsAlloc((size_t)G * 4);
    float*    den   = (float*)   wsAlloc((size_t)G * 4);
    float*    bnA   = (float*)   wsAlloc((size_t)L * H * 4);
    float*    bnB   = (float*)   wsAlloc((size_t)L * H * 4);
    float*    vnA   = (float*)   wsAlloc((size_t)(L - 1) * H * 4);
    float*    vnB   = (float*)   wsAlloc((size_t)(L - 1) * H * 4);
    _Float16* nodeT = (_Float16*)wsAlloc((size_t)NP304 * KN * 2);
    _Float16* edgeT = (_Float16*)wsAlloc((size_t)NP304 * KE * 2);
    _Float16* w1T   = (_Float16*)wsAlloc((size_t)L * NP304 * HP * 2);
    _Float16* w2T   = (_Float16*)wsAlloc((size_t)L * NP304 * HP * 2);
    _Float16* vw1T  = (_Float16*)wsAlloc((size_t)(L - 1) * NP304 * HP * 2);
    _Float16* vw2T  = (_Float16*)wsAlloc((size_t)(L - 1) * NP304 * HP * 2);
    _Float16* attT  = (_Float16*)wsAlloc((size_t)ATTHP * HP * 2);
    _Float16* projT = (_Float16*)wsAlloc((size_t)OUTD * HP * 2);
    // big reused scratch: [agg f32 N*H][zbuf f16 N*HP][tbuf f16 N*HP]
    size_t aggB = (size_t)N * H * 4, zB = (size_t)N * HP * 2, tB = (size_t)N * HP * 2;
    char* scr = (char*)wsAlloc(aggB + zB + tB);
    float*    agg  = (float*)scr;
    _Float16* zbuf = (_Float16*)(scr + aggB);
    _Float16* tbuf = (_Float16*)(scr + aggB + zB);
    _Float16* Xn   = (_Float16*)scr;                    // phase A (N*KN*2 <= aggB)
    _Float16* Xe   = (_Float16*)scr;                    // phase A (after node GEMM)
    _Float16* s1   = (_Float16*)scr;                    // phase C (N*ATTHP*2 <= aggB)
    _Float16* h16  = zbuf;                              // phase C
    // carve inside tbuf slot (dead when these are live)
    char* ts = (char*)tbuf;
    size_t toff = 0;
    auto tAlloc = [&](size_t bytes) -> void* {
        void* p = ts + toff;
        toff += (bytes + 255) & ~(size_t)255;
        return p;
    };
    _Float16* vtbuf  = (_Float16*)tAlloc((size_t)G * HP * 2);
    _Float16* vhid   = (_Float16*)tAlloc((size_t)G * HP * 2);
    float*    gpool  = (float*)   tAlloc((size_t)G * H * 4);
    _Float16* gbuf   = (_Float16*)tAlloc((size_t)G * HP * 2);
    float*    outpre = (float*)   tAlloc((size_t)G * OUTD * 4);

    auto blk = [](size_t total) -> unsigned { return (unsigned)((total + 255) / 256); };
    auto gemm = [&](const _Float16* A, int lda, const _Float16* BT, int ldb,
                    const float* bias, const float* bna, const float* bnb,
                    const float* res, int ldr,
                    float* Cf, _Float16* Ch, int ldc,
                    int M, int Nout, int Kp, int mode) {
        int tilesM = (M + 15) / 16, tilesN = (Nout + 15) / 16;
        long tiles = (long)tilesM * tilesN;
        unsigned blocks = (unsigned)((tiles + 7) / 8);   // 8 waves per 256-thread block
        wmma_gemm<<<blocks, 256, 0, stream>>>(A, lda, BT, ldb, bias, bna, bnb,
                                              res, ldr, Cf, Ch, ldc,
                                              M, Nout, Kp, tilesM, tilesN, mode);
    };

    // ---- pack weights (transposed, padded, f16) + fold BN ----
    pack_wT<<<blk((size_t)NP304 * KN), 256, 0, stream>>>(node_proj_w, nodeT, 9 * NEW, H, KN, NP304);
    pack_wT<<<blk((size_t)NP304 * KE), 256, 0, stream>>>(edge_proj_w, edgeT, 3 * NEW, H, KE, NP304);
    for (int i = 0; i < L; ++i) {
        pack_wT<<<blk((size_t)NP304 * HP), 256, 0, stream>>>(conv_w1 + (size_t)i * H * H, w1T + (size_t)i * NP304 * HP, H, H, HP, NP304);
        pack_wT<<<blk((size_t)NP304 * HP), 256, 0, stream>>>(conv_w2 + (size_t)i * H * H, w2T + (size_t)i * NP304 * HP, H, H, HP, NP304);
    }
    for (int i = 0; i < L - 1; ++i) {
        pack_wT<<<blk((size_t)NP304 * HP), 256, 0, stream>>>(vn_w1 + (size_t)i * H * H, vw1T + (size_t)i * NP304 * HP, H, H, HP, NP304);
        pack_wT<<<blk((size_t)NP304 * HP), 256, 0, stream>>>(vn_w2 + (size_t)i * H * H, vw2T + (size_t)i * NP304 * HP, H, H, HP, NP304);
    }
    pack_wT<<<blk((size_t)ATTHP * HP), 256, 0, stream>>>(att_w1, attT, H, ATTH, HP, ATTHP);
    pack_wT<<<blk((size_t)OUTD * HP), 256, 0, stream>>>(proj_w, projT, H, OUTD, HP, OUTD);
    bn_prep<<<blk((size_t)L * H), 256, 0, stream>>>(bn_g, bn_b, bn_m, bn_v, bnA, bnB, L * H);
    bn_prep<<<blk((size_t)(L - 1) * H), 256, 0, stream>>>(vn_bn_g, vn_bn_b, vn_bn_m, vn_bn_v, vnA, vnB, (L - 1) * H);

    // ---- embeddings -> h, efeat ----
    node_gather<<<blk((size_t)N * KN), 256, 0, stream>>>(x, nt[0], nt[1], nt[2], nt[3], nt[4], nt[5], nt[6], nt[7], nt[8], Xn, N);
    gemm(Xn, KN, nodeT, KN, node_proj_b, nullptr, nullptr, nullptr, 0, h, nullptr, H, N, H, KN, 0);
    edge_gather<<<blk((size_t)E * KE), 256, 0, stream>>>(eattr, et[0], et[1], et[2], Xe, E);
    gemm(Xe, KE, edgeT, KE, edge_proj_b, nullptr, nullptr, nullptr, 0, nullptr, efeat, H, E, H, KE, 4);
    init_vh<<<blk((size_t)G * H), 256, 0, stream>>>(vh, v_emb, G);

    // ---- GINE layers ----
    for (int i = 0; i < L; ++i) {
        add_vnode<<<blk((size_t)N * H), 256, 0, stream>>>(h, vh, batch, N);
        fill_f32<<<blk((size_t)N * H), 256, 0, stream>>>(agg, 0.0f, (size_t)N * H);
        msg_scatter<<<blk((size_t)E * H), 256, 0, stream>>>(h, efeat, src, dst, agg, E);
        to_half_pad<<<blk((size_t)N * HP), 256, 0, stream>>>(h, agg, zbuf, N);
        gemm(zbuf, HP, w1T + (size_t)i * NP304 * HP, HP, conv_b1 + (size_t)i * H,
             nullptr, nullptr, nullptr, 0, nullptr, tbuf, HP, N, H, HP, 1);
        gemm(tbuf, HP, w2T + (size_t)i * NP304 * HP, HP, conv_b2 + (size_t)i * H,
             bnA + (size_t)i * H, bnB + (size_t)i * H, h, H, h, nullptr, H, N, H, HP, 3);
        if (i < L - 1) {
            fill_f32<<<blk((size_t)G * H), 256, 0, stream>>>(vt, 0.0f, (size_t)G * H);
            seg_sum_h<<<blk((size_t)N * H), 256, 0, stream>>>(h, batch, vt, N);
            to_half_pad<<<blk((size_t)G * HP), 256, 0, stream>>>(vt, vh, vtbuf, G);
            gemm(vtbuf, HP, vw1T + (size_t)i * NP304 * HP, HP, vn_b1 + (size_t)i * H,
                 vnA + (size_t)i * H, vnB + (size_t)i * H, nullptr, 0,
                 nullptr, vhid, HP, G, H, HP, 2);
            gemm(vhid, HP, vw2T + (size_t)i * NP304 * HP, HP, vn_b2 + (size_t)i * H,
                 nullptr, nullptr, nullptr, 0, vh, nullptr, H, G, H, HP, 0);
        }
    }

    // ---- attention softmax pooling + projection + L2 normalize ----
    to_half_pad<<<blk((size_t)N * HP), 256, 0, stream>>>(h, nullptr, h16, N);
    gemm(h16, HP, attT, HP, att_b1, nullptr, nullptr, nullptr, 0, nullptr, s1, ATTHP, N, ATTH, HP, 1);
    att_dot<<<blk((size_t)N), 256, 0, stream>>>(s1, att_w2, att_b2, sb, N);
    fill_f32<<<blk((size_t)G), 256, 0, stream>>>(smax, -INFINITY, (size_t)G);
    fill_f32<<<blk((size_t)G), 256, 0, stream>>>(den, 0.0f, (size_t)G);
    seg_max<<<blk((size_t)N), 256, 0, stream>>>(sb, batch, smax, N);
    exp_den<<<blk((size_t)N), 256, 0, stream>>>(sb, batch, smax, exb, den, N);
    fill_f32<<<blk((size_t)G * H), 256, 0, stream>>>(gpool, 0.0f, (size_t)G * H);
    pool_weighted<<<blk((size_t)N * H), 256, 0, stream>>>(h, exb, den, batch, gpool, N);
    to_half_pad<<<blk((size_t)G * HP), 256, 0, stream>>>(gpool, nullptr, gbuf, G);
    gemm(gbuf, HP, projT, HP, proj_b, nullptr, nullptr, nullptr, 0, outpre, nullptr, OUTD, G, OUTD, HP, 0);
    normalize_rows<<<G, 256, 0, stream>>>(outpre, (float*)d_out, OUTD);
}
